// MMDLoss_8040178778143
// MI455X (gfx1250) — compile-verified, matrix operand order checked by reading, not run on hardware
//
#include <hip/hip_runtime.h>
#include <stdint.h>

typedef __attribute__((ext_vector_type(16))) __bf16 v16bf;
typedef __attribute__((ext_vector_type(8)))  float  v8f;

#define N_TOT 8192
#define N_SRC 4096
#define DIM   1024
#define BT    128
#define NTILE 64   // 8192 / 128

// d_ws layout (bytes)
static constexpr size_t OFF_HI  = 0;
static constexpr size_t OFF_LO  = (size_t)N_TOT * DIM * 2;            // 16 MB
static constexpr size_t OFF_SQ  = OFF_LO + (size_t)N_TOT * DIM * 2;   // 32 MB
static constexpr size_t OFF_ACC = OFF_SQ + (size_t)N_TOT * 4;         // +32 KB (8B aligned)

union FragB { v16bf v; uint4 q[2]; };
union FragC { v8f  v; float f[8]; };

__device__ __forceinline__ unsigned short f32_to_bf16_rne(float x) {
    uint32_t u = __float_as_uint(x);
    uint32_t r = u + 0x7FFFu + ((u >> 16) & 1u);
    return (unsigned short)(r >> 16);
}

// ---- Pass A: fp32 -> (hi,lo) bf16 split + row squared-norms + zero accumulators ----
__global__ void mmd_prep_kernel(const float* __restrict__ src, const float* __restrict__ tgt,
                                unsigned short* __restrict__ hi, unsigned short* __restrict__ lo,
                                float* __restrict__ sq, double* __restrict__ acc) {
    int row = blockIdx.x;
    int t   = threadIdx.x;
    const float* x = (row < N_SRC) ? (src + (size_t)row * DIM)
                                   : (tgt + (size_t)(row - N_SRC) * DIM);
    size_t base = (size_t)row * DIM;
    float s = 0.0f;
    for (int i = t; i < DIM; i += 256) {
        float v = x[i];
        s += v * v;
        unsigned short hb = f32_to_bf16_rne(v);
        float hf = __uint_as_float((uint32_t)hb << 16);
        unsigned short lb = f32_to_bf16_rne(v - hf);
        hi[base + i] = hb;
        lo[base + i] = lb;
    }
    __shared__ float red[256];
    red[t] = s;
    __syncthreads();
    for (int off = 128; off > 0; off >>= 1) {
        if (t < off) red[t] += red[t + off];
        __syncthreads();
    }
    if (t == 0) {
        sq[row] = red[0];
        if (row == 0) { acc[0] = 0.0; acc[1] = 0.0; }
    }
}

// ---- bandwidth from sum(l2) ----
__global__ void mmd_bw_kernel(const double* __restrict__ acc, float* __restrict__ cparam) {
    double denom = (double)N_TOT * (double)N_TOT - (double)N_TOT;
    double bw = acc[0] / denom / 4.0;              // / KERNEL_MUL^(KERNEL_NUM//2)
    cparam[0] = (float)(1.0 / (bw * 16.0));        // exponent scale for e = exp(-l2*c)
}

// ---- Gram-matrix tile + l2 epilogue (mode 0: sum l2, mode 1: signed kernel sum) ----
// 128 threads = 4 waves (2x2); wave tile 64x64 = 4x4 WMMA accumulators.
// Per K-step (K=32): 32 global_load_b128 feed 48 v_wmma (bf16x3 split).
__global__ __launch_bounds__(128)
void mmd_gemm_kernel(const unsigned short* __restrict__ hi,
                     const unsigned short* __restrict__ lo,
                     const float* __restrict__ sq,
                     double* __restrict__ acc,
                     const float* __restrict__ cparam,
                     int mode) {
    // upper-triangular tile mapping (square 128x128 tiles; transpose tile has equal sum)
    int rem = blockIdx.x, tm = 0;
    while (rem >= NTILE - tm) { rem -= NTILE - tm; ++tm; }
    int tn = tm + rem;
    float w = (tm == tn) ? 1.0f : 2.0f;

    int tid  = threadIdx.x;
    int wave = tid >> 5;
    int lane = tid & 31;
    int wm = wave & 1;            // 2 row groups x 64 rows
    int wn = wave >> 1;           // 2 col groups x 64 cols
    int m16   = lane & 15;
    int khalf = (lane >> 4) << 3; // 0 or 8 (ISA 16-bit A/B fragment K packing)

    int rowbase = tm * BT + wm * 64;
    int colbase = tn * BT + wn * 64;

    // per-fragment element offsets (K part added in the loop)
    size_t rowA[4], rowB[4];
#pragma unroll
    for (int i = 0; i < 4; ++i) {
        rowA[i] = (size_t)(rowbase + i * 16 + m16) * DIM + khalf;
        rowB[i] = (size_t)(colbase + i * 16 + m16) * DIM + khalf;
    }

    FragC accv[4][4];
#pragma unroll
    for (int mi = 0; mi < 4; ++mi)
#pragma unroll
        for (int ni = 0; ni < 4; ++ni)
#pragma unroll
            for (int r = 0; r < 8; ++r) accv[mi][ni].f[r] = 0.0f;

    for (int kk = 0; kk < DIM; kk += 32) {
        FragB aH[4], aL[4], bH[4], bL[4];
#pragma unroll
        for (int i = 0; i < 4; ++i) {
            size_t ra = rowA[i] + kk;
            aH[i].q[0] = *(const uint4*)(hi + ra);
            aH[i].q[1] = *(const uint4*)(hi + ra + 16);
            aL[i].q[0] = *(const uint4*)(lo + ra);
            aL[i].q[1] = *(const uint4*)(lo + ra + 16);
            size_t rb = rowB[i] + kk;
            bH[i].q[0] = *(const uint4*)(hi + rb);
            bH[i].q[1] = *(const uint4*)(hi + rb + 16);
            bL[i].q[0] = *(const uint4*)(lo + rb);
            bL[i].q[1] = *(const uint4*)(lo + rb + 16);
        }
#pragma unroll
        for (int mi = 0; mi < 4; ++mi)
#pragma unroll
            for (int ni = 0; ni < 4; ++ni) {
                accv[mi][ni].v = __builtin_amdgcn_wmma_f32_16x16x32_bf16(
                    false, aH[mi].v, false, bH[ni].v, (short)0, accv[mi][ni].v, false, false);
                accv[mi][ni].v = __builtin_amdgcn_wmma_f32_16x16x32_bf16(
                    false, aH[mi].v, false, bL[ni].v, (short)0, accv[mi][ni].v, false, false);
                accv[mi][ni].v = __builtin_amdgcn_wmma_f32_16x16x32_bf16(
                    false, aL[mi].v, false, bH[ni].v, (short)0, accv[mi][ni].v, false, false);
            }
    }

    // epilogue: l2 = relu(sq_m + sq_n - 2G); reduce
    float c = (mode == 1) ? cparam[0] : 0.0f;
    int nIdx  = lane & 15;
    int mhalf = (lane >> 4) * 8;
    double lsum = 0.0;
#pragma unroll
    for (int mi = 0; mi < 4; ++mi) {
#pragma unroll
        for (int ni = 0; ni < 4; ++ni) {
            int gn = colbase + ni * 16 + nIdx;
            float sqn = sq[gn];
            float sgncol = (gn < N_SRC) ? 1.0f : -1.0f;
#pragma unroll
            for (int r = 0; r < 8; ++r) {
                int gm = rowbase + mi * 16 + mhalf + r;
                float l2 = fmaxf(sq[gm] + sqn - 2.0f * accv[mi][ni].f[r], 0.0f);
                if (mode == 0) {
                    lsum += (double)l2;
                } else {
                    float e  = __expf(-l2 * c);
                    float e2 = e * e, e4 = e2 * e2, e8 = e4 * e4, e16 = e8 * e8;
                    float k  = e + e2 + e4 + e8 + e16;
                    float sgn = ((gm < N_SRC) ? 1.0f : -1.0f) * sgncol;
                    lsum += (double)(sgn * k);
                }
            }
        }
    }
    __shared__ double red[128];
    red[tid] = lsum;
    __syncthreads();
    for (int off = 64; off > 0; off >>= 1) {
        if (tid < off) red[tid] += red[tid + off];
        __syncthreads();
    }
    if (tid == 0) atomicAdd(&acc[mode], (double)w * red[0]);
}

__global__ void mmd_final_kernel(const double* __restrict__ acc, float* __restrict__ out) {
    out[0] = (float)(acc[1] / ((double)N_SRC * (double)N_SRC));
}

extern "C" void kernel_launch(void* const* d_in, const int* in_sizes, int n_in,
                              void* d_out, int out_size, void* d_ws, size_t ws_size,
                              hipStream_t stream) {
    const float* src = (const float*)d_in[0];
    const float* tgt = (const float*)d_in[1];
    char* ws = (char*)d_ws;
    unsigned short* hi = (unsigned short*)(ws + OFF_HI);
    unsigned short* lo = (unsigned short*)(ws + OFF_LO);
    float*  sq     = (float*)(ws + OFF_SQ);
    double* acc    = (double*)(ws + OFF_ACC);
    float*  cparam = (float*)(ws + OFF_ACC + 16);

    mmd_prep_kernel<<<N_TOT, 256, 0, stream>>>(src, tgt, hi, lo, sq, acc);
    int nblocks = NTILE * (NTILE + 1) / 2;  // 2080 upper-triangular tiles
    mmd_gemm_kernel<<<nblocks, 128, 0, stream>>>(hi, lo, sq, acc, cparam, 0);
    mmd_bw_kernel<<<1, 1, 0, stream>>>(acc, cparam);
    mmd_gemm_kernel<<<nblocks, 128, 0, stream>>>(hi, lo, sq, acc, cparam, 1);
    mmd_final_kernel<<<1, 1, 0, stream>>>(acc, (float*)d_out);
}
